// GAT_3083786519242
// MI455X (gfx1250) — compile-verified, matrix operand order checked by reading, not run on hardware
//
#include <hip/hip_runtime.h>
#include <math.h>
#include <stdint.h>

typedef __attribute__((ext_vector_type(16))) _Float16 v16h;
typedef __attribute__((ext_vector_type(8)))  _Float16 v8h;
typedef __attribute__((ext_vector_type(4)))  _Float16 v4h;
typedef __attribute__((ext_vector_type(8)))  float    v8f;

#define NEG_SLOPE 0.2f

// ---------------------------------------------------------------------------
// fp32 -> f16 bulk convert (4 elems/thread, packed cvt + b64 store)
// ---------------------------------------------------------------------------
__global__ void cvt_a_kernel(const float* __restrict__ s,
                             _Float16* __restrict__ d, long long n4) {
  const long long i = (long long)blockIdx.x * blockDim.x + threadIdx.x;
  if (i >= n4) return;
  const float4 f = ((const float4*)s)[i];
  v4h h;
  h[0] = (_Float16)f.x; h[1] = (_Float16)f.y;
  h[2] = (_Float16)f.z; h[3] = (_Float16)f.w;
  ((v4h*)d)[i] = h;
}

// W[K x NC] fp32  ->  Wt[NCpad x K] f16, zero-padded columns (NC..NCpad)
__global__ void cvt_w_kernel(const float* __restrict__ W,
                             _Float16* __restrict__ Wt,
                             int K, int NC, int NCpad) {
  const int idx = blockIdx.x * blockDim.x + threadIdx.x;
  if (idx >= NCpad * K) return;
  const int n = idx / K, k = idx - n * K;
  Wt[idx] = (n < NC) ? (_Float16)W[(size_t)k * NC + n] : (_Float16)0.0f;
}

// ============================================================================
// WMMA GEMM: C[N x NC] = Ah[N x K](f16) * Wt[NCpad x K](f16, transposed)
// Block = 256 thr = 8 waves; one 16-col tile per block, staged in LDS via
// GLOBAL_LOAD_ASYNC_TO_LDS_B128 (ASYNCcnt), 8 row tiles (one per wave).
// Fragment layout per ISA 7.12.2: a lane's v16h = two contiguous 8-half
// chunks at K = k0+8*half and k0+16+8*half.
//   -> A: 2x global_load_b128 ; B: 2x ds_load_b128 ; then v_wmma.
// ============================================================================
template <int K>
__global__ void wmma_gemm_f16_kernel(const _Float16* __restrict__ Ah,
                                     const _Float16* __restrict__ Wt,
                                     float* __restrict__ C,
                                     int Nrows, int NC, int NCpad) {
  __shared__ __align__(16) _Float16 sB[16 * K];
  const int tilesN = NCpad >> 4;
  const int tn   = blockIdx.x % tilesN;
  const int brow = blockIdx.x / tilesN;          // group of 8 row tiles
  const int n0   = tn << 4;

  // stage the 16 x K f16 B tile directly into LDS: one async b128 per thread
  {
    const int nChunks = (16 * K) / 8;            // 16-byte chunks
    if ((int)threadIdx.x < nChunks) {
      const _Float16* g = Wt + (size_t)n0 * K + (size_t)threadIdx.x * 8;
      const unsigned lds = (unsigned)(uintptr_t)sB + threadIdx.x * 16u;
      asm volatile("global_load_async_to_lds_b128 %0, %1, off"
                   :: "v"(lds), "v"(g) : "memory");
    }
    asm volatile("s_wait_asynccnt 0" ::: "memory");
  }
  __syncthreads();

  const int wavei = threadIdx.x >> 5;
  const int tm = brow * 8 + wavei;
  if (tm >= (Nrows >> 4)) return;
  const int lane = threadIdx.x & 31;
  const int half = lane >> 4, r = lane & 15;

  const _Float16* Ap = Ah + (size_t)((tm << 4) + r) * K;  // lane's A row
  const _Float16* Bp = sB + r * K;                        // lane's B column

  v8f acc = {};
#pragma unroll
  for (int k0 = 0; k0 < K; k0 += 32) {
    union { v16h v; v8h h[2]; } a, b;
    a.h[0] = *(const v8h*)(Ap + k0 + 8 * half);
    a.h[1] = *(const v8h*)(Ap + k0 + 16 + 8 * half);
    b.h[0] = *(const v8h*)(Bp + k0 + 8 * half);
    b.h[1] = *(const v8h*)(Bp + k0 + 16 + 8 * half);
    // 8 args: (neg_a, A, neg_b, B, c_mod, C, reuse_a, reuse_b)
    acc = __builtin_amdgcn_wmma_f32_16x16x32_f16(false, a.v, false, b.v,
                                                 (short)0, acc, false, false);
  }
  const int n = n0 + r;
  if (n < NC) {
#pragma unroll
    for (int rr = 0; rr < 8; ++rr)
      C[(size_t)((tm << 4) + rr + 8 * half) * NC + n] = acc[rr];
  }
}

// ---------------------------------------------------------------------------
// el/er: per-(node,head) dot of feat with attention vectors
// ---------------------------------------------------------------------------
template <int H, int D>
__global__ void attn_kernel(const float* __restrict__ feat,
                            const float* __restrict__ al,
                            const float* __restrict__ ar,
                            float* __restrict__ el, float* __restrict__ er,
                            int Nn) {
  const int idx = blockIdx.x * blockDim.x + threadIdx.x;
  if (idx >= Nn * H) return;
  const int i = idx / H, h = idx - i * H;
  const float* f = feat + (size_t)i * H * D + (size_t)h * D;
  float sl = 0.f, sr = 0.f;
#pragma unroll
  for (int d = 0; d < D; ++d) {
    sl += f[d] * al[h * D + d];
    sr += f[d] * ar[h * D + d];
  }
  el[idx] = sl;
  er[idx] = sr;
}

__global__ void fill_kernel(float* __restrict__ p, float v, size_t n) {
  const size_t i = (size_t)blockIdx.x * blockDim.x + threadIdx.x;
  if (i < n) p[i] = v;
}

// float atomic max via sign-split integer atomics (global_atomic_max/min)
__device__ __forceinline__ void atomicMaxF(float* addr, float v) {
  if (v >= 0.0f) atomicMax((int*)addr, __float_as_int(v));
  else           atomicMin((unsigned int*)addr, __float_as_uint(v));
}

template <int H>
__global__ void edge_max_kernel(const int* __restrict__ src,
                                const int* __restrict__ dst,
                                const float* __restrict__ el,
                                const float* __restrict__ er,
                                float* __restrict__ emax, long long E) {
  const long long idx = (long long)blockIdx.x * blockDim.x + threadIdx.x;
  if (idx >= E * H) return;
  const long long e = idx / H;
  const int h = (int)(idx - e * H);
  const int s = src[e], d = dst[e];
  float v = el[(size_t)s * H + h] + er[(size_t)d * H + h];
  v = (v > 0.f) ? v : NEG_SLOPE * v;
  atomicMaxF(&emax[(size_t)d * H + h], v);
}

template <int H>
__global__ void edge_expsum_kernel(const int* __restrict__ src,
                                   const int* __restrict__ dst,
                                   const float* __restrict__ el,
                                   const float* __restrict__ er,
                                   const float* __restrict__ emax,
                                   float* __restrict__ ee,
                                   float* __restrict__ denom, long long E) {
  const long long idx = (long long)blockIdx.x * blockDim.x + threadIdx.x;
  if (idx >= E * H) return;
  const long long e = idx / H;
  const int h = (int)(idx - e * H);
  const int s = src[e], d = dst[e];
  float v = el[(size_t)s * H + h] + er[(size_t)d * H + h];
  v = (v > 0.f) ? v : NEG_SLOPE * v;
  const float x = __expf(v - emax[(size_t)d * H + h]);
  ee[idx] = x;
  atomicAdd(&denom[(size_t)d * H + h], x);
}

// Dominant kernel: one thread per (edge, channel); consecutive threads cover
// the HD contiguous channels of one edge -> coalesced gather + atomic scatter.
template <int H, int D>
__global__ void edge_scatter_kernel(const int* __restrict__ src,
                                    const int* __restrict__ dst,
                                    const float* __restrict__ feat,
                                    const float* __restrict__ ee,
                                    const float* __restrict__ denom,
                                    float* __restrict__ rst, long long E) {
  constexpr int HD = H * D;
  const long long idx = (long long)blockIdx.x * blockDim.x + threadIdx.x;
  if (idx >= E * HD) return;
  const long long e = idx / HD;
  const int c = (int)(idx - e * HD);
  const int h = c / D;
  const int s = src[e], d = dst[e];
  const float alpha = ee[e * H + h] / denom[(size_t)d * H + h];
  atomicAdd(&rst[(size_t)d * HD + c], feat[(size_t)s * HD + c] * alpha);
}

template <int HD>
__global__ void finish_kernel(const float* __restrict__ rst,
                              const float* __restrict__ res,   // may be null
                              const float* __restrict__ bias,
                              float* __restrict__ out, int Nn, int act) {
  const long long idx = (long long)blockIdx.x * blockDim.x + threadIdx.x;
  if (idx >= (long long)Nn * HD) return;
  const int c = (int)(idx % HD);
  float v = rst[idx] + bias[c];
  if (res) v += res[idx];
  if (act) v = fmaxf(v, 0.0f);
  out[idx] = v;
}

template <int H>
__global__ void mean_kernel(const float* __restrict__ h, float* __restrict__ out,
                            int Nn) {
  const int i = blockIdx.x * blockDim.x + threadIdx.x;
  if (i >= Nn) return;
  float s = 0.f;
#pragma unroll
  for (int k = 0; k < H; ++k) s += h[(size_t)i * H + k];
  out[i] = s * (1.0f / (float)H);
}

// ============================================================================
// Host-side layer driver (stream launches only; graph-capture safe)
// ============================================================================
struct Bufs {
  float *feat, *res, *el, *er, *emax, *denom, *ee, *rst, *hbuf;
  _Float16 *Ah, *WtA, *WtB;
  const int *src, *dst;
  int N;
  long long E;
};

template <int K, int H, int D>
static void run_layer(const Bufs& B, const float* hin,
                      const float* W, const float* al, const float* ar,
                      const float* bias, const float* resW, int act,
                      float* hout, hipStream_t stream) {
  constexpr int HD = H * D;
  constexpr int NCpad = ((HD + 15) / 16) * 16;
  constexpr int tilesN = NCpad / 16;
  const int N = B.N;
  const long long E = B.E;

  // f16 staging: activations + transposed/padded weights
  const long long n4 = (long long)N * K / 4;
  cvt_a_kernel<<<(int)((n4 + 255) / 256), 256, 0, stream>>>(hin, B.Ah, n4);
  cvt_w_kernel<<<(NCpad * K + 255) / 256, 256, 0, stream>>>(W, B.WtA, K, HD, NCpad);
  if (resW)
    cvt_w_kernel<<<(NCpad * K + 255) / 256, 256, 0, stream>>>(resW, B.WtB, K, HD, NCpad);

  const int rowGroups = ((N >> 4) + 7) / 8;          // 8 row tiles per block
  wmma_gemm_f16_kernel<K><<<rowGroups * tilesN, 256, 0, stream>>>(
      B.Ah, B.WtA, B.feat, N, HD, NCpad);
  if (resW)
    wmma_gemm_f16_kernel<K><<<rowGroups * tilesN, 256, 0, stream>>>(
        B.Ah, B.WtB, B.res, N, HD, NCpad);

  attn_kernel<H, D><<<(N * H + 255) / 256, 256, 0, stream>>>(B.feat, al, ar,
                                                             B.el, B.er, N);
  {
    const size_t nh = (size_t)N * H, nhd = (size_t)N * HD;
    fill_kernel<<<(int)((nh + 255) / 256), 256, 0, stream>>>(B.emax, -INFINITY, nh);
    fill_kernel<<<(int)((nh + 255) / 256), 256, 0, stream>>>(B.denom, 0.0f, nh);
    fill_kernel<<<(int)((nhd + 255) / 256), 256, 0, stream>>>(B.rst, 0.0f, nhd);
  }
  const long long eh = E * H, ehd = E * HD;
  edge_max_kernel<H><<<(int)((eh + 255) / 256), 256, 0, stream>>>(
      B.src, B.dst, B.el, B.er, B.emax, E);
  edge_expsum_kernel<H><<<(int)((eh + 255) / 256), 256, 0, stream>>>(
      B.src, B.dst, B.el, B.er, B.emax, B.ee, B.denom, E);
  edge_scatter_kernel<H, D><<<(int)((ehd + 255) / 256), 256, 0, stream>>>(
      B.src, B.dst, B.feat, B.ee, B.denom, B.rst, E);
  finish_kernel<HD><<<(int)(((long long)N * HD + 255) / 256), 256, 0, stream>>>(
      B.rst, resW ? B.res : nullptr, bias, hout, N, act);
}

extern "C" void kernel_launch(void* const* d_in, const int* in_sizes, int n_in,
                              void* d_out, int out_size, void* d_ws, size_t ws_size,
                              hipStream_t stream) {
  const float* x    = (const float*)d_in[0];
  const int*   src  = (const int*)d_in[1];
  const int*   dst  = (const int*)d_in[2];
  const float* W1   = (const float*)d_in[3];
  const float* al1  = (const float*)d_in[4];
  const float* ar1  = (const float*)d_in[5];
  const float* b1   = (const float*)d_in[6];
  const float* W2   = (const float*)d_in[7];
  const float* al2  = (const float*)d_in[8];
  const float* ar2  = (const float*)d_in[9];
  const float* b2   = (const float*)d_in[10];
  const float* rW2  = (const float*)d_in[11];
  const float* W3   = (const float*)d_in[12];
  const float* al3  = (const float*)d_in[13];
  const float* ar3  = (const float*)d_in[14];
  const float* b3   = (const float*)d_in[15];
  const float* rW3  = (const float*)d_in[16];

  const int       N = in_sizes[0] / 64;     // 50000
  const long long E = in_sizes[1];          // 850000

  // workspace layout
  float* p = (float*)d_ws;
  Bufs B;
  B.feat  = p;  p += (size_t)N * 128;
  B.res   = p;  p += (size_t)N * 128;
  B.el    = p;  p += (size_t)N * 8;
  B.er    = p;  p += (size_t)N * 8;
  B.emax  = p;  p += (size_t)N * 8;
  B.denom = p;  p += (size_t)N * 8;
  B.ee    = p;  p += (size_t)E * 6;
  B.rst   = p;  p += (size_t)N * 128;
  B.hbuf  = p;  p += (size_t)N * 128;
  B.Ah    = (_Float16*)p;  p += (size_t)N * 64;   // N*128 halves
  B.WtA   = (_Float16*)p;  p += 8192;             // 128*128 halves
  B.WtB   = (_Float16*)p;  p += 8192;
  B.src = src; B.dst = dst; B.N = N; B.E = E;

  // layer 1: 64 -> 4x32, relu, no residual        (reads x, writes hbuf)
  run_layer<64, 4, 32>(B, x, W1, al1, ar1, b1, nullptr, 1, B.hbuf, stream);
  // layer 2: 128 -> 4x32, relu, residual          (hbuf -> hbuf; stream-ordered)
  run_layer<128, 4, 32>(B, B.hbuf, W2, al2, ar2, b2, rW2, 1, B.hbuf, stream);
  // layer 3: 128 -> 6x1, no act, residual
  run_layer<128, 6, 1>(B, B.hbuf, W3, al3, ar3, b3, rW3, 0, B.hbuf, stream);
  // mean over 6 heads -> (N,)
  mean_kernel<6><<<(N + 255) / 256, 256, 0, stream>>>(B.hbuf, (float*)d_out, N);
}
